// Network_57638461112712
// MI455X (gfx1250) — matrix operand order repaired and checked by reading, not compiled, not run on hardware
//
#include <hip/hip_runtime.h>
#include <hip/hip_bf16.h>
#include <math.h>

typedef __attribute__((ext_vector_type(16))) _Float16 v16h;
typedef __attribute__((ext_vector_type(8)))  float    v8f;

#define C_S 0.3826834323650898f
#define C_X 0.9238795325112867f
#define INV4 0.25f            /* 1/sqrt(16), also 1/sqrt(NUM_NEIGHBORS) */
#define INV_SQRT10 0.31622776601683794f
#define INV8 0.125f           /* 1/sqrt(64) */
#define INV_SQRT32 0.17677669529663687f
#define INV_SQRT48 0.14433756729740643f
#define INV_SQRT3 0.5773502691896258f

// Fast silu/sigmoid: v_exp_f32 + v_rcp_f32 (no IEEE div Newton chains).
__device__ __forceinline__ float siluf(float x) {
  return x * __builtin_amdgcn_rcpf(1.0f + __expf(-x));
}
__device__ __forceinline__ float sigmf(float x) {
  return __builtin_amdgcn_rcpf(1.0f + __expf(-x));
}

// ---------- WMMA fragment helpers (wave32, 16x16x32 f16) ----------
// A 16x32 f16: lanes L and L+16 hold row M=L%16; half i -> K = ((i<8)?i:i+8) + (lane>>4)*8
// B 32x16 f16: lane L holds column N=L%16; half i -> K = ((i<8)?i:i+8) + (lane>>4)*8
// C/D 16x16 f32: v8f, element r -> M = r + 8*(lane>>4), N = lane%16

__device__ __forceinline__ v16h load_b_frag(const float* __restrict__ W, int Kreal,
                                            int Ncols, int ct, int kbase, float scale,
                                            int lane) {
  int col  = (lane & 15) + ct * 16;
  int ksec = (lane >> 4) & 1;
  v16h b;
#pragma unroll
  for (int i = 0; i < 16; ++i) {
    int K = kbase + ((i < 8) ? i : (i + 8)) + ksec * 8;
    float v = (K < Kreal) ? W[K * Ncols + col] * scale : 0.0f;
    b[i] = (_Float16)v;
  }
  return b;
}

// Build an A fragment from an LDS row of f16 (row stride passed by caller).
__device__ __forceinline__ v16h load_a_lds(const _Float16* __restrict__ hrow,
                                           int kbase, int khalf) {
  v16h a;
#pragma unroll
  for (int i = 0; i < 16; ++i) {
    int base = (i < 8) ? i : (i + 8);   // contiguous runs of 8 -> ds_load_b128 pairs
    a[i] = hrow[kbase + base + khalf * 8];
  }
  return a;
}

#define WMMA_F16(a, b, c) \
  __builtin_amdgcn_wmma_f32_16x16x32_f16(false, (a), false, (b), (short)0, (c), false, false)

#define WAVE_DS_FENCE() asm volatile("s_wait_dscnt 0x0" ::: "memory")

// ============================================================
// Kernel 1: x1 = node_input @ W_lin1_1 * 1/4 * na
// ============================================================
__global__ void node_pass1(const float* __restrict__ ni, const float* __restrict__ na,
                           const float* __restrict__ W, float* __restrict__ x1, int N) {
  int n = blockIdx.x * blockDim.x + threadIdx.x;
  if (n >= N) return;
  float in[16];
#pragma unroll
  for (int k = 0; k < 16; ++k) in[k] = ni[(size_t)n * 16 + k];
  float s = na[n] * INV4;
#pragma unroll
  for (int j = 0; j < 16; ++j) {
    float acc = 0.0f;
#pragma unroll
    for (int k = 0; k < 16; ++k) acc = fmaf(in[k], W[k * 16 + j], acc);
    x1[(size_t)n * 16 + j] = acc * s;
  }
}

// ============================================================
// Kernel 2 (edge pass 1): radial1 on WMMA + message + atomic segment-sum
// ============================================================
__global__ void __launch_bounds__(128, 1)
edge_pass1(const float* __restrict__ elem, const float* __restrict__ ea,
           const int* __restrict__ esrc, const int* __restrict__ edst,
           const float* __restrict__ x1, const float* __restrict__ W1,
           const float* __restrict__ W2, float* __restrict__ mid, int E) {
  __shared__ _Float16 sh_e[4][16 * 32];  // staged elem tile (K padded to 32)
  __shared__ _Float16 sh_h[4][16 * 64];  // layer-1 activations
  __shared__ float    sh_w[4][16 * 32];  // radial output

  const int wave = threadIdx.x >> 5;
  const int lane = threadIdx.x & 31;
  const int row = lane & 15, khalf = lane >> 4, col16 = lane & 15;

  // Zero the elem staging buffer once; the K>=10 pad is never overwritten.
#pragma unroll
  for (int t = 0; t < 16; ++t) sh_e[wave][lane + 32 * t] = (_Float16)0.0f;

  // Weight fragments live in VGPRs across the whole loop.
  v16h bW1[4];
#pragma unroll
  for (int ct = 0; ct < 4; ++ct) bW1[ct] = load_b_frag(W1, 10, 64, ct, 0, 1.0f, lane);
  v16h bW2[2][2];  // [kchunk][ct], 1/8 folded in
#pragma unroll
  for (int kc = 0; kc < 2; ++kc)
#pragma unroll
    for (int ct = 0; ct < 2; ++ct) bW2[kc][ct] = load_b_frag(W2, 64, 32, ct, kc * 32, INV8, lane);
  WAVE_DS_FENCE();

  const int tiles = (E + 15) / 16;
  const int gwave = blockIdx.x * 4 + wave;
  const int nwaves = gridDim.x * 4;

  for (int tile = gwave; tile < tiles; tile += nwaves) {
    const int e0 = tile * 16;

    // ---- branchless coalesced stage of 16x10 elem rows into padded 16x32 f16 ----
    // Row index clamped to E-1: tail-tile garbage rows feed WMMA rows whose
    // outputs are never scattered, so an unconditional load/store is safe.
#pragma unroll
    for (int it = 0; it < 5; ++it) {
      int t = lane + 32 * it;                 // 0..159
      int r = t / 10, K = t - r * 10;
      long long ge = e0 + r;
      long long gec = (ge < E) ? ge : (long long)(E - 1);
      sh_e[wave][r * 32 + K] = (_Float16)(elem[gec * 10 + K] * INV_SQRT10);
    }
    WAVE_DS_FENCE();

    // ---- layer 1: h_pre = elem @ W1 / sqrt(10) ----
    v16h a = load_a_lds(&sh_e[wave][row * 32], 0, khalf);
    v8f acc[4];
#pragma unroll
    for (int t = 0; t < 4; ++t) { v8f z = {}; acc[t] = WMMA_F16(a, bW1[t], z); }

    // silu, stash h (f16) to LDS so we can rebuild the layer-2 A fragment
#pragma unroll
    for (int t = 0; t < 4; ++t)
#pragma unroll
      for (int r = 0; r < 8; ++r)
        sh_h[wave][(r + 8 * khalf) * 64 + t * 16 + col16] = (_Float16)siluf(acc[t][r]);
    WAVE_DS_FENCE();

    v16h ah0 = load_a_lds(&sh_h[wave][row * 64], 0,  khalf);
    v16h ah1 = load_a_lds(&sh_h[wave][row * 64], 32, khalf);

    // ---- layer 2: w = h @ (W2/8), 16x32 output ----
    v8f acc2[2];
#pragma unroll
    for (int ct = 0; ct < 2; ++ct) {
      v8f z = {};
      z = WMMA_F16(ah0, bW2[0][ct], z);
      acc2[ct] = WMMA_F16(ah1, bW2[1][ct], z);
    }
#pragma unroll
    for (int ct = 0; ct < 2; ++ct)
#pragma unroll
      for (int r = 0; r < 8; ++r)
        sh_w[wave][(r + 8 * khalf) * 32 + ct * 16 + col16] = acc2[ct][r];
    WAVE_DS_FENCE();

    // ---- message + scatter: ef[64] per edge, atomic into mid[dst] ----
    if (e0 + 16 <= E) {
      // fast path: full tile, no per-edge guard
#pragma unroll 1
      for (int le = 0; le < 16; ++le) {
        const int ge = e0 + le;
        const int src = esrc[ge], dst = edst[ge];
        const float a0 = ea[(size_t)ge * 4 + 0];
#pragma unroll
        for (int hh = 0; hh < 2; ++hh) {
          const int j = lane + 32 * hh;
          float val;
          if (j < 16) {
            val = x1[(size_t)src * 16 + j] * a0 * sh_w[wave][le * 32 + j];
          } else {
            const int k = (j - 16) / 3, c = (j - 16) % 3;
            val = x1[(size_t)src * 16 + k] * sh_w[wave][le * 32 + 16 + k] *
                  ea[(size_t)ge * 4 + 1 + c];
          }
          atomicAdd(&mid[(size_t)dst * 64 + j], val * INV4);
        }
      }
    } else {
#pragma unroll 1
      for (int le = 0; le < 16; ++le) {
        const int ge = e0 + le;
        if (ge >= E) break;
        const int src = esrc[ge], dst = edst[ge];
        const float a0 = ea[(size_t)ge * 4 + 0];
#pragma unroll
        for (int hh = 0; hh < 2; ++hh) {
          const int j = lane + 32 * hh;
          float val;
          if (j < 16) {
            val = x1[(size_t)src * 16 + j] * a0 * sh_w[wave][le * 32 + j];
          } else {
            const int k = (j - 16) / 3, c = (j - 16) % 3;
            val = x1[(size_t)src * 16 + k] * sh_w[wave][le * 32 + 16 + k] *
                  ea[(size_t)ge * 4 + 1 + c];
          }
          atomicAdd(&mid[(size_t)dst * 64 + j], val * INV4);
        }
      }
    }
  }
}

// ============================================================
// Kernel 3: node nonlinearity + gating -> h0, hv, C_S*s2 into out
// ============================================================
__global__ void node_pass2(const float* __restrict__ ni, const float* __restrict__ naP,
                           const float* __restrict__ mid, const float* __restrict__ W_sc1,
                           const float* __restrict__ W_l2s, const float* __restrict__ W_l2v,
                           const float* __restrict__ W_sc2, const float* __restrict__ W12s,
                           const float* __restrict__ W12v, float* __restrict__ h0,
                           float* __restrict__ hv, float* __restrict__ out, int N) {
  int n = blockIdx.x * blockDim.x + threadIdx.x;
  if (n >= N) return;
  const float na = naP[n];
  float in[16], m[64];
#pragma unroll
  for (int k = 0; k < 16; ++k) in[k] = ni[(size_t)n * 16 + k];
#pragma unroll
  for (int j = 0; j < 64; ++j) m[j] = mid[(size_t)n * 64 + j];

  float gs[48];
#pragma unroll
  for (int j = 0; j < 48; ++j) {
    float ssc = 0.0f, ys = 0.0f;
#pragma unroll
    for (int k = 0; k < 16; ++k) {
      ssc = fmaf(in[k], W_sc1[k * 48 + j], ssc);
      ys  = fmaf(m[k],  W_l2s[k * 48 + j], ys);
    }
    gs[j] = INV4 * na * (C_S * ssc + C_X * ys);
  }
  float scal[32], gates[16];
#pragma unroll
  for (int j = 0; j < 32; ++j) scal[j] = siluf(gs[j]);
#pragma unroll
  for (int k = 0; k < 16; ++k) gates[k] = sigmf(gs[32 + k]);

  float vec[48];
#pragma unroll
  for (int u = 0; u < 16; ++u)
#pragma unroll
    for (int c = 0; c < 3; ++c) {
      float yv = 0.0f;
#pragma unroll
      for (int p = 0; p < 16; ++p) yv = fmaf(m[16 + p * 3 + c], W_l2v[p * 16 + u], yv);
      vec[u * 3 + c] = INV4 * na * yv * gates[u];
    }

  float s2 = 0.0f;
#pragma unroll
  for (int j = 0; j < 32; ++j) s2 = fmaf(scal[j], W_sc2[j], s2);
  out[n] = C_S * (s2 * INV_SQRT32 * na);

#pragma unroll
  for (int j = 0; j < 32; ++j) {
    float acc = 0.0f;
#pragma unroll
    for (int k = 0; k < 32; ++k) acc = fmaf(scal[k], W12s[k * 32 + j], acc);
    h0[(size_t)n * 32 + j] = acc * INV_SQRT32 * na;
  }
#pragma unroll
  for (int w = 0; w < 16; ++w)
#pragma unroll
    for (int c = 0; c < 3; ++c) {
      float acc = 0.0f;
#pragma unroll
      for (int u = 0; u < 16; ++u) acc = fmaf(vec[u * 3 + c], W12v[u * 16 + w], acc);
      hv[(size_t)n * 48 + w * 3 + c] = acc * INV4 * na;
    }
}

// ============================================================
// Kernel 4 (edge pass 2): radial2 on WMMA + message + atomic segment-sum
// ============================================================
__global__ void __launch_bounds__(128, 1)
edge_pass2(const float* __restrict__ elem, const float* __restrict__ ea,
           const int* __restrict__ esrc, const int* __restrict__ edst,
           const float* __restrict__ h0, const float* __restrict__ hv,
           const float* __restrict__ W1, const float* __restrict__ W2,
           float* __restrict__ mid2, int E) {
  __shared__ _Float16 sh_e[4][16 * 32];
  __shared__ _Float16 sh_h[4][16 * 64];
  __shared__ float    sh_w[4][16 * 48];

  const int wave = threadIdx.x >> 5;
  const int lane = threadIdx.x & 31;
  const int row = lane & 15, khalf = lane >> 4, col16 = lane & 15;

#pragma unroll
  for (int t = 0; t < 16; ++t) sh_e[wave][lane + 32 * t] = (_Float16)0.0f;

  v16h bW1[4];
#pragma unroll
  for (int ct = 0; ct < 4; ++ct) bW1[ct] = load_b_frag(W1, 10, 64, ct, 0, 1.0f, lane);
  v16h bW2[2][3];  // [kchunk][ct], 1/8 folded in; output width 48
#pragma unroll
  for (int kc = 0; kc < 2; ++kc)
#pragma unroll
    for (int ct = 0; ct < 3; ++ct) bW2[kc][ct] = load_b_frag(W2, 64, 48, ct, kc * 32, INV8, lane);
  WAVE_DS_FENCE();

  const int tiles = (E + 15) / 16;
  const int gwave = blockIdx.x * 4 + wave;
  const int nwaves = gridDim.x * 4;

  for (int tile = gwave; tile < tiles; tile += nwaves) {
    const int e0 = tile * 16;

#pragma unroll
    for (int it = 0; it < 5; ++it) {
      int t = lane + 32 * it;
      int r = t / 10, K = t - r * 10;
      long long ge = e0 + r;
      long long gec = (ge < E) ? ge : (long long)(E - 1);
      sh_e[wave][r * 32 + K] = (_Float16)(elem[gec * 10 + K] * INV_SQRT10);
    }
    WAVE_DS_FENCE();

    v16h a = load_a_lds(&sh_e[wave][row * 32], 0, khalf);
    v8f acc[4];
#pragma unroll
    for (int t = 0; t < 4; ++t) { v8f z = {}; acc[t] = WMMA_F16(a, bW1[t], z); }
#pragma unroll
    for (int t = 0; t < 4; ++t)
#pragma unroll
      for (int r = 0; r < 8; ++r)
        sh_h[wave][(r + 8 * khalf) * 64 + t * 16 + col16] = (_Float16)siluf(acc[t][r]);
    WAVE_DS_FENCE();

    v16h ah0 = load_a_lds(&sh_h[wave][row * 64], 0,  khalf);
    v16h ah1 = load_a_lds(&sh_h[wave][row * 64], 32, khalf);

    v8f acc2[3];
#pragma unroll
    for (int ct = 0; ct < 3; ++ct) {
      v8f z = {};
      z = WMMA_F16(ah0, bW2[0][ct], z);
      acc2[ct] = WMMA_F16(ah1, bW2[1][ct], z);
    }
#pragma unroll
    for (int ct = 0; ct < 3; ++ct)
#pragma unroll
      for (int r = 0; r < 8; ++r)
        sh_w[wave][(r + 8 * khalf) * 48 + ct * 16 + col16] = acc2[ct][r];
    WAVE_DS_FENCE();

    if (e0 + 16 <= E) {
#pragma unroll 1
      for (int le = 0; le < 16; ++le) {
        const int ge = e0 + le;
        const int src = esrc[ge], dst = edst[ge];
        const float a0  = ea[(size_t)ge * 4 + 0];
        const float a1x = ea[(size_t)ge * 4 + 1];
        const float a1y = ea[(size_t)ge * 4 + 2];
        const float a1z = ea[(size_t)ge * 4 + 3];
        {
          const int j = lane;  // fa: j = 0..31
          float val = h0[(size_t)src * 32 + j] * a0 * sh_w[wave][le * 48 + j];
          atomicAdd(&mid2[(size_t)dst * 48 + j], val * INV4);
        }
        if (lane < 16) {       // fb: k = 0..15
          const int k = lane;
          float dot = hv[(size_t)src * 48 + k * 3 + 0] * a1x +
                      hv[(size_t)src * 48 + k * 3 + 1] * a1y +
                      hv[(size_t)src * 48 + k * 3 + 2] * a1z;
          float val = dot * INV_SQRT3 * sh_w[wave][le * 48 + 32 + k];
          atomicAdd(&mid2[(size_t)dst * 48 + 32 + k], val * INV4);
        }
      }
    } else {
#pragma unroll 1
      for (int le = 0; le < 16; ++le) {
        const int ge = e0 + le;
        if (ge >= E) break;
        const int src = esrc[ge], dst = edst[ge];
        const float a0  = ea[(size_t)ge * 4 + 0];
        const float a1x = ea[(size_t)ge * 4 + 1];
        const float a1y = ea[(size_t)ge * 4 + 2];
        const float a1z = ea[(size_t)ge * 4 + 3];
        {
          const int j = lane;
          float val = h0[(size_t)src * 32 + j] * a0 * sh_w[wave][le * 48 + j];
          atomicAdd(&mid2[(size_t)dst * 48 + j], val * INV4);
        }
        if (lane < 16) {
          const int k = lane;
          float dot = hv[(size_t)src * 48 + k * 3 + 0] * a1x +
                      hv[(size_t)src * 48 + k * 3 + 1] * a1y +
                      hv[(size_t)src * 48 + k * 3 + 2] * a1z;
          float val = dot * INV_SQRT3 * sh_w[wave][le * 48 + 32 + k];
          atomicAdd(&mid2[(size_t)dst * 48 + 32 + k], val * INV4);
        }
      }
    }
  }
}

// ============================================================
// Kernel 5: out += C_X * (mid2 @ W_lin2_2) / sqrt(48) * na
// ============================================================
__global__ void node_pass3(const float* __restrict__ naP, const float* __restrict__ mid2,
                           const float* __restrict__ Wl2, float* __restrict__ out, int N) {
  int n = blockIdx.x * blockDim.x + threadIdx.x;
  if (n >= N) return;
  float acc = 0.0f;
#pragma unroll
  for (int j = 0; j < 48; ++j) acc = fmaf(mid2[(size_t)n * 48 + j], Wl2[j], acc);
  out[n] = out[n] + C_X * (acc * INV_SQRT48 * naP[n]);
}

extern "C" void kernel_launch(void* const* d_in, const int* in_sizes, int n_in,
                              void* d_out, int out_size, void* d_ws, size_t ws_size,
                              hipStream_t stream) {
  const float* node_input = (const float*)d_in[0];
  const float* node_attr  = (const float*)d_in[1];
  const float* edge_attr  = (const float*)d_in[2];
  const float* elem       = (const float*)d_in[3];
  const float* W_sc1      = (const float*)d_in[4];
  const float* W_lin1_1   = (const float*)d_in[5];
  const float* fc1_W1     = (const float*)d_in[6];
  const float* fc1_W2     = (const float*)d_in[7];
  const float* W_l2s_1    = (const float*)d_in[8];
  const float* W_l2v_1    = (const float*)d_in[9];
  const float* W_sc2      = (const float*)d_in[10];
  const float* W_lin1_2s  = (const float*)d_in[11];
  const float* W_lin1_2v  = (const float*)d_in[12];
  const float* fc2_W1     = (const float*)d_in[13];
  const float* fc2_W2     = (const float*)d_in[14];
  const float* W_lin2_2   = (const float*)d_in[15];
  const int*   esrc       = (const int*)d_in[16];
  const int*   edst       = (const int*)d_in[17];

  const int N = in_sizes[0] / 16;
  const int E = in_sizes[16];

  float* ws   = (float*)d_ws;
  float* x1   = ws;                       // N*16
  float* mid  = x1  + (size_t)N * 16;     // N*64
  float* h0   = mid + (size_t)N * 64;     // N*32
  float* hv   = h0  + (size_t)N * 32;     // N*48
  float* mid2 = hv  + (size_t)N * 48;     // N*48

  hipMemsetAsync(mid,  0, (size_t)N * 64 * sizeof(float), stream);
  hipMemsetAsync(mid2, 0, (size_t)N * 48 * sizeof(float), stream);

  const int nb = (N + 255) / 256;
  node_pass1<<<nb, 256, 0, stream>>>(node_input, node_attr, W_lin1_1, x1, N);
  edge_pass1<<<4096, 128, 0, stream>>>(elem, edge_attr, esrc, edst, x1,
                                       fc1_W1, fc1_W2, mid, E);
  node_pass2<<<nb, 256, 0, stream>>>(node_input, node_attr, mid, W_sc1, W_l2s_1,
                                     W_l2v_1, W_sc2, W_lin1_2s, W_lin1_2v,
                                     h0, hv, (float*)d_out, N);
  edge_pass2<<<4096, 128, 0, stream>>>(elem, edge_attr, esrc, edst, h0, hv,
                                       fc2_W1, fc2_W2, mid2, E);
  node_pass3<<<nb, 256, 0, stream>>>(node_attr, mid2, W_lin2_2, (float*)d_out, N);
}
